// GCN_309237645608
// MI455X (gfx1250) — matrix operand order repaired and checked by reading, not compiled, not run on hardware
//
#include <hip/hip_runtime.h>

typedef __attribute__((ext_vector_type(16))) __bf16 v16bf;
typedef __attribute__((ext_vector_type(8)))  float  v8f;

#define N_NODES  50000
#define N_EDGES  600000
#define N_GRAPHS 64
#define IN_C     128
#define HID_C    128
#define OUT_C    64

// ---- f32 -> bf16 round-to-nearest-even ----
__device__ __forceinline__ __bf16 f2bf(float f) {
    unsigned int u = __builtin_bit_cast(unsigned int, f);
    u += 0x7FFFu + ((u >> 16) & 1u);
    unsigned short s = (unsigned short)(u >> 16);
    return __builtin_bit_cast(__bf16, s);
}

// ================= degree / normalization =================
__global__ void __launch_bounds__(256) k_deg_init(float* deg) {
    int i = blockIdx.x * 256 + threadIdx.x;
    if (i < N_NODES) deg[i] = 1.0f;              // self-loop contribution
}

__global__ void __launch_bounds__(256) k_deg_edges(const int* __restrict__ ei, float* deg) {
    int e = blockIdx.x * 256 + threadIdx.x;
    if (e < N_EDGES) atomicAdd(&deg[ei[N_EDGES + e]], 1.0f);   // dst side
}

__global__ void __launch_bounds__(256) k_dinv(float* deg) {
    int i = blockIdx.x * 256 + threadIdx.x;
    if (i < N_NODES) deg[i] = rsqrtf(fmaxf(deg[i], 1.0f));
}

// ================= f32 -> bf16 bulk convert =================
__global__ void __launch_bounds__(256) k_cvt_bf16(const float* __restrict__ src,
                                                  __bf16* __restrict__ dst, int n) {
    int t = blockIdx.x * 256 + threadIdx.x;
    if (t < n) dst[t] = f2bf(src[t]);
}

// ===== GEMM: Out[N,COUT] = A[N,128](bf16) @ W[128,COUT](f32), bf16 WMMA =====
// A tile staged via global_load_async_to_lds_b128 (ASYNCcnt), W pre-swizzled
// into the exact B-fragment layout so each lane reads 32 contiguous bytes.
template<int COUT>
__global__ void __launch_bounds__(256) k_gemm_wmma(
        const __bf16* __restrict__ A, const float* __restrict__ W,
        float* __restrict__ Out, int n_rows) {
    __shared__ __bf16 As[128 * 128];                 // 32 KB, row-major tile
    __shared__ __bf16 Wf[(COUT / 4) * 32 * 16];      // fragments: [frag][lane][16]

    const int tid  = threadIdx.x;
    const int row0 = blockIdx.x * 128;
    const bool full = (row0 + 128 <= n_rows);        // block-uniform

    // ---- W -> fragment-swizzled LDS (ISA 7.12.2 B layout: 32x16 bf16) ----
    for (int i = tid; i < 128 * COUT; i += 256) {
        int k = i / COUT, c = i % COUT;              // W row-major [k][c]
        int ct = c >> 4, n = c & 15;
        int kc = k >> 5, krem = k & 31;
        int lane = (((krem >> 3) & 1) << 4) | n;     // K bit 3 -> lanes 16..31
        int ii = (krem & 7) | ((krem >> 4) << 3);    // element within fragment
        Wf[(((ct << 2) | kc) * 32 + lane) * 16 + ii] = f2bf(W[i]);
    }

    // ---- A tile: async bulk copy (full tiles) or guarded tail fill ----
    if (full) {
        unsigned lbase = (unsigned)(size_t)&As[0];   // low 32 bits = LDS offset
        const char* g = (const char*)(A + (size_t)row0 * 128);
        #pragma unroll
        for (int it = 0; it < 8; ++it) {             // 256 thr * 8 * 16B = 32 KB
            int t16 = tid + it * 256;
            unsigned loff = lbase + t16 * 16;
            const char* ga = g + (size_t)t16 * 16;
            asm volatile("global_load_async_to_lds_b128 %0, %1, off"
                         :: "v"(loff), "v"(ga) : "memory");
        }
        asm volatile("s_wait_asynccnt 0x0" ::: "memory");
    } else {
        for (int i = tid; i < 128 * 128; i += 256) {
            int r = i >> 7, c = i & 127;
            int gr = row0 + r;
            As[i] = (gr < n_rows) ? A[(size_t)gr * 128 + c]
                                  : __builtin_bit_cast(__bf16, (unsigned short)0);
        }
    }
    __syncthreads();

    const int wave = tid >> 5;
    const int lane = tid & 31;
    const int m    = lane & 15;
    const int hi   = lane >> 4;
    const int rt   = wave * 16;                      // 8 waves cover 128 rows

    // ---- preload A fragments once (reused across all column tiles) ----
    v16bf afrag[4];
    #pragma unroll
    for (int kc = 0; kc < 4; ++kc) {
        const int abase = (rt + m) * 128 + kc * 32 + hi * 8;
        #pragma unroll
        for (int i = 0; i < 8; ++i) {
            afrag[kc][i]     = As[abase + i];        // K = 32kc + 8hi + i
            afrag[kc][i + 8] = As[abase + 16 + i];   // K = 32kc + 8hi + 16 + i
        }
    }

    const int rbase = row0 + rt + hi * 8;            // first row this lane stores

    for (int ct = 0; ct < COUT / 16; ++ct) {
        v8f acc = {};
        #pragma unroll
        for (int kc = 0; kc < 4; ++kc) {
            v16bf b = *(const v16bf*)&Wf[(((ct << 2) | kc) * 32 + lane) * 16];
            acc = __builtin_amdgcn_wmma_f32_16x16x32_bf16(
                false, afrag[kc], false, b, (short)0, acc, false, false);
        }
        // D: VGPR i -> M = i + 8hi, N = m
        float* outp = Out + (size_t)rbase * COUT + ct * 16 + m;
        if (full) {                                  // uniform branch, no exec churn
            #pragma unroll
            for (int i = 0; i < 8; ++i)
                outp[(size_t)i * COUT] = acc[i];
        } else {
            #pragma unroll
            for (int i = 0; i < 8; ++i)
                if (rbase + i < n_rows)
                    outp[(size_t)i * COUT] = acc[i];
        }
    }
}

// ============== aggregation: init with self-loop term (also zero-fills) ==============
template<int C>
__global__ void __launch_bounds__(256) k_agg_init(
        const float* __restrict__ h, const float* __restrict__ dinv, float* __restrict__ a) {
    int t = blockIdx.x * 256 + threadIdx.x;
    if (t < N_NODES * C) {
        float d = dinv[t / C];
        a[t] = d * d * h[t];
    }
}

// wave-per-edge scatter-add: a[dst] += h[src] * dinv[src]*dinv[dst]
template<int C>
__global__ void __launch_bounds__(256) k_agg_edges(
        const int* __restrict__ ei, const float* __restrict__ h,
        const float* __restrict__ dinv, float* __restrict__ a) {
    int e = blockIdx.x * 8 + (threadIdx.x >> 5);
    if (e >= N_EDGES) return;
    int lane = threadIdx.x & 31;
    int src = ei[e];
    int dst = ei[N_EDGES + e];
    float nrm = dinv[src] * dinv[dst];
    if constexpr (C == 128) {
        const float4 hv = *(const float4*)(h + (size_t)src * C + lane * 4);  // b128 gather
        float* ap = a + (size_t)dst * C + lane * 4;
        atomicAdd(ap + 0, hv.x * nrm);
        atomicAdd(ap + 1, hv.y * nrm);
        atomicAdd(ap + 2, hv.z * nrm);
        atomicAdd(ap + 3, hv.w * nrm);
    } else {
        const float2 hv = *(const float2*)(h + (size_t)src * C + lane * 2);  // b64 gather
        float* ap = a + (size_t)dst * C + lane * 2;
        atomicAdd(ap + 0, hv.x * nrm);
        atomicAdd(ap + 1, hv.y * nrm);
    }
}

// layer-1 epilogue: bf16(relu(a + b)) -> GEMM2 input
template<int C>
__global__ void __launch_bounds__(256) k_bias_relu_bf16(
        const float* __restrict__ a, const float* __restrict__ b, __bf16* __restrict__ o) {
    int t = blockIdx.x * 256 + threadIdx.x;
    if (t < N_NODES * C)
        o[t] = f2bf(fmaxf(a[t] + b[t & (C - 1)], 0.0f));
}

// layer-2 epilogue: a += b (f32, in place)
template<int C>
__global__ void __launch_bounds__(256) k_bias_add(float* a, const float* __restrict__ b) {
    int t = blockIdx.x * 256 + threadIdx.x;
    if (t < N_NODES * C) a[t] += b[t & (C - 1)];
}

// ================= global mean pool =================
__global__ void __launch_bounds__(256) k_pool_zero(float* sums, float* cnt) {
    int t = blockIdx.x * 256 + threadIdx.x;
    if (t < N_GRAPHS * OUT_C) sums[t] = 0.0f;
    if (t < N_GRAPHS) cnt[t] = 0.0f;
}

__global__ void __launch_bounds__(256) k_pool_acc(
        const float* __restrict__ a, const int* __restrict__ batch,
        float* sums, float* cnt) {
    int t = blockIdx.x * 256 + threadIdx.x;
    if (t < N_NODES * OUT_C) {
        int n = t >> 6, c = t & 63;
        int g = batch[n];
        atomicAdd(&sums[g * OUT_C + c], a[t]);
        if (c == 0) atomicAdd(&cnt[g], 1.0f);
    }
}

__global__ void __launch_bounds__(256) k_pool_norm(
        const float* __restrict__ sums, const float* __restrict__ cnt, float* out) {
    int t = blockIdx.x * 256 + threadIdx.x;
    if (t < N_GRAPHS * OUT_C) out[t] = sums[t] / fmaxf(cnt[t >> 6], 1.0f);
}

// ================= driver =================
extern "C" void kernel_launch(void* const* d_in, const int* in_sizes, int n_in,
                              void* d_out, int out_size, void* d_ws, size_t ws_size,
                              hipStream_t stream) {
    const float* x  = (const float*)d_in[0];   // [50000,128]
    const float* W1 = (const float*)d_in[1];   // [128,128]
    const float* b1 = (const float*)d_in[2];   // [128]
    const float* W2 = (const float*)d_in[3];   // [128,64]
    const float* b2 = (const float*)d_in[4];   // [64]
    const int*   ei = (const int*)d_in[5];     // [2,600000]
    const int*   bt = (const int*)d_in[6];     // [50000]
    float* out = (float*)d_out;                // [64,64]

    char* ws = (char*)d_ws;
    const size_t MB = 1024 * 1024;
    float*  dinv = (float*)ws;                         // 50000 f32
    __bf16* BX   = (__bf16*)(ws + 1 * MB);             // 12.8 MB: xbf, then a1bf
    float*  BH   = (float*)(ws + 14 * MB);             // 25.6 MB: h1, then h2
    float*  BA   = (float*)(ws + 40 * MB);             // 25.6 MB: a1, then a2
    float*  psum = (float*)(ws + 66 * MB);             // 64*64 f32
    float*  pcnt = psum + N_GRAPHS * OUT_C;            // 64 f32

    __bf16* xbf  = BX;
    float*  h1   = BH;
    float*  a1   = BA;
    __bf16* a1bf = BX;   // reuse: xbf dead after GEMM1
    float*  h2   = BH;   // reuse: h1 dead after layer-1 edge aggregation
    float*  a2   = BA;   // reuse: a1 dead after layer-1 epilogue

    const int TB = 256;
    // degrees -> dinv
    k_deg_init <<<(N_NODES + TB - 1) / TB, TB, 0, stream>>>(dinv);
    k_deg_edges<<<(N_EDGES + TB - 1) / TB, TB, 0, stream>>>(ei, dinv);
    k_dinv     <<<(N_NODES + TB - 1) / TB, TB, 0, stream>>>(dinv);

    // layer 1: h1 = x @ W1 ; a1 = norm-agg(h1) ; a1bf = bf16(relu(a1 + b1))
    k_cvt_bf16 <<<(N_NODES * IN_C + TB - 1) / TB, TB, 0, stream>>>(x, xbf, N_NODES * IN_C);
    k_gemm_wmma<HID_C><<<(N_NODES + 127) / 128, TB, 0, stream>>>(xbf, W1, h1, N_NODES);
    k_agg_init <HID_C><<<(N_NODES * HID_C + TB - 1) / TB, TB, 0, stream>>>(h1, dinv, a1);
    k_agg_edges<HID_C><<<(N_EDGES + 7) / 8, TB, 0, stream>>>(ei, h1, dinv, a1);
    k_bias_relu_bf16<HID_C><<<(N_NODES * HID_C + TB - 1) / TB, TB, 0, stream>>>(a1, b1, a1bf);

    // layer 2: h2 = a1 @ W2 ; a2 = norm-agg(h2) ; a2 += b2
    k_gemm_wmma<OUT_C><<<(N_NODES + 127) / 128, TB, 0, stream>>>(a1bf, W2, h2, N_NODES);
    k_agg_init <OUT_C><<<(N_NODES * OUT_C + TB - 1) / TB, TB, 0, stream>>>(h2, dinv, a2);
    k_agg_edges<OUT_C><<<(N_EDGES + 7) / 8, TB, 0, stream>>>(ei, h2, dinv, a2);
    k_bias_add <OUT_C><<<(N_NODES * OUT_C + TB - 1) / TB, TB, 0, stream>>>(a2, b2);

    // mean pool
    k_pool_zero<<<(N_GRAPHS * OUT_C + TB - 1) / TB, TB, 0, stream>>>(psum, pcnt);
    k_pool_acc <<<(N_NODES * OUT_C + TB - 1) / TB, TB, 0, stream>>>(a2, bt, psum, pcnt);
    k_pool_norm<<<(N_GRAPHS * OUT_C + TB - 1) / TB, TB, 0, stream>>>(psum, pcnt, out);
}
